// Pointer_33973191311660
// MI455X (gfx1250) — matrix operand order repaired and checked by reading, not compiled, not run on hardware
//
#include <hip/hip_runtime.h>
#include <hip/hip_bf16.h>

// ---------------------------------------------------------------------------
// Problem constants (from reference): B=4, T=128, S=512, Dq=Dk=128, U=128
// ---------------------------------------------------------------------------
#define BATCH 4
#define TQ    128
#define SK    512
#define DK    128
#define UNITS 128

typedef __attribute__((ext_vector_type(16))) _Float16 v16h;
typedef __attribute__((ext_vector_type(8)))  float    v8f;
typedef __attribute__((ext_vector_type(4)))  int      v4i;

// Hardware tanh on gfx1250 (V_TANH_F32), fallback to libm expansion.
#if __has_builtin(__builtin_amdgcn_tanhf)
static __device__ __forceinline__ float fast_tanh(float x) { return __builtin_amdgcn_tanhf(x); }
#else
static __device__ __forceinline__ float fast_tanh(float x) { return tanhf(x); }
#endif

// CDNA5 async global->LDS copy path (ASYNCcnt-tracked), guarded.
#if defined(__AMDGCN__) && __has_builtin(__builtin_amdgcn_global_load_async_to_lds_b128) && __has_builtin(__builtin_amdgcn_s_wait_asynccnt)
#define HAVE_ASYNC_LDS 1
#else
#define HAVE_ASYNC_LDS 0
#endif

// ---------------------------------------------------------------------------
// Kernel 1: out[M,128] = A[M,128] @ W[128,128] + bias[128]
// One wave32 per 16x16 output tile. K marched in 4 steps of 32 using
// v_wmma_f32_16x16x32_f16 (f16 operands, f32 accumulate). Launch geometry is
// exact (M multiple of 16, tiles multiple of waves/block) so EXEC is all-1s.
//
// Operand layouts (CDNA5 ISA 7.12.2, 16-bit data, wave32):
//  A 16x32 : lane half h=lane>>4, m=lane&15
//            elem e<8  -> K = 8h + e
//            elem e>=8 -> K = 16 + 8h + (e-8)
//  B 32x16 : n = lane&15; elem e<8 -> K = 8h + e, e>=8 -> K = 16 + 8h + (e-8)
//  C/D 16x16 f32: VGPR r -> (M = r + 8h, N = lane&15)
// ---------------------------------------------------------------------------
__global__ void gemm_bias_wmma(const float* __restrict__ A,
                               const float* __restrict__ W,
                               const float* __restrict__ bias,
                               float* __restrict__ out,
                               int M)
{
    const int K = 128, N = 128;
    const int lane    = threadIdx.x & 31;
    const int wave    = threadIdx.x >> 5;
    const int tilesN  = N / 16;                       // 8
    const int tileId  = blockIdx.x * (blockDim.x >> 5) + wave;
    const int tileM   = tileId / tilesN;
    const int tileN   = tileId % tilesN;
    const int half    = lane >> 4;                    // 0 or 1
    const int l15     = lane & 15;

    const int rowA = tileM * 16 + l15;                // A row this lane feeds
    const int colB = tileN * 16 + l15;                // B/D column this lane feeds

    (void)M;

    // Bias: every C/D element this lane holds has N = colB.
    v8f acc;
    const float bv = bias[colB];
#pragma unroll
    for (int r = 0; r < 8; ++r) acc[r] = bv;

    const float* __restrict__ arow = A + (size_t)rowA * K;

    for (int k0 = 0; k0 < K; k0 += 32) {
        const int ka = k0 + (half << 3);              // A: +8*half
        v16h a;
#pragma unroll
        for (int e = 0; e < 8; ++e) a[e]     = (_Float16)arow[ka + e];
#pragma unroll
        for (int e = 0; e < 8; ++e) a[8 + e] = (_Float16)arow[ka + 16 + e];

        const int kb = k0 + (half << 3);              // B: +8*half
        v16h b;
#pragma unroll
        for (int e = 0; e < 16; ++e) b[e] = (_Float16)W[(size_t)(kb + e + (e >= 8 ? 8 : 0)) * N + colB];

        acc = __builtin_amdgcn_wmma_f32_16x16x32_f16(
            /*neg_a=*/false, a, /*neg_b=*/false, b,
            /*c_mod=*/(short)0, acc, /*reuse_a=*/false, /*reuse_b=*/false);
    }

    const int baseRow = tileM * 16 + (half << 3);
#pragma unroll
    for (int r = 0; r < 8; ++r)
        out[(size_t)(baseRow + r) * N + colB] = acc[r];
}

// ---------------------------------------------------------------------------
// Kernel 2: score[b,t,s] = bV + sum_u V[u] * tanh(enc[b,s,u] + dec[b,t,u])
// Block = 256 threads, tile = 16 t x 16 s, one score per thread.
// enc/dec tiles staged in LDS with +4 float row padding (stride 132 floats:
// s*132 % 64 banks = 4s -> conflict-free ds_load_b128 across the half-wave).
// ---------------------------------------------------------------------------
__global__ void score_tanh_kernel(const float* __restrict__ enc,   // [B*S, U]
                                  const float* __restrict__ dec,   // [B*T, U]
                                  const float* __restrict__ Vw,    // [U]
                                  const float* __restrict__ bV,    // [1]
                                  float* __restrict__ out)         // [B*T*S]
{
    __shared__ __align__(16) float sEnc[16][UNITS + 4];
    __shared__ __align__(16) float sDec[16][UNITS + 4];
    __shared__ __align__(16) float sV[UNITS];

    const int tid = threadIdx.x;
    const int b   = blockIdx.z;
    const int t0  = blockIdx.y * 16;
    const int s0  = blockIdx.x * 16;

    const float* __restrict__ encB = enc + ((size_t)b * SK + s0) * UNITS;
    const float* __restrict__ decB = dec + ((size_t)b * TQ + t0) * UNITS;

    // Cooperative tile load: 16 rows x 128 floats; each thread moves 8 floats.
    const int lrow = tid >> 4;              // 0..15
    const int lcol = (tid & 15) * 8;        // 0,8,...,120
    const float* ge = encB + lrow * UNITS + lcol;
    const float* gd = decB + lrow * UNITS + lcol;

#if HAVE_ASYNC_LDS
    __builtin_amdgcn_global_load_async_to_lds_b128(
        (v4i*)ge,       (v4i*)&sEnc[lrow][lcol],     0, 0);
    __builtin_amdgcn_global_load_async_to_lds_b128(
        (v4i*)(ge + 4), (v4i*)&sEnc[lrow][lcol + 4], 0, 0);
    __builtin_amdgcn_global_load_async_to_lds_b128(
        (v4i*)gd,       (v4i*)&sDec[lrow][lcol],     0, 0);
    __builtin_amdgcn_global_load_async_to_lds_b128(
        (v4i*)(gd + 4), (v4i*)&sDec[lrow][lcol + 4], 0, 0);
#else
    *(float4*)&sEnc[lrow][lcol]     = *(const float4*)ge;
    *(float4*)&sEnc[lrow][lcol + 4] = *(const float4*)(ge + 4);
    *(float4*)&sDec[lrow][lcol]     = *(const float4*)gd;
    *(float4*)&sDec[lrow][lcol + 4] = *(const float4*)(gd + 4);
#endif
    if (tid < UNITS) sV[tid] = Vw[tid];

#if HAVE_ASYNC_LDS
    __builtin_amdgcn_s_wait_asynccnt(0);
#endif
    __syncthreads();

    const int t = tid >> 4;   // 0..15
    const int s = tid & 15;   // 0..15

    float acc = bV[0];
    const float* er = &sEnc[s][0];
    const float* dr = &sDec[t][0];
#pragma unroll
    for (int u = 0; u < UNITS; u += 4) {
        const float4 e = *(const float4*)(er + u);
        const float4 d = *(const float4*)(dr + u);
        const float4 v = *(const float4*)(&sV[u]);
        acc += v.x * fast_tanh(e.x + d.x);
        acc += v.y * fast_tanh(e.y + d.y);
        acc += v.z * fast_tanh(e.z + d.z);
        acc += v.w * fast_tanh(e.w + d.w);
    }

    out[((size_t)b * TQ + (t0 + t)) * SK + (s0 + s)] = acc;
}

// ---------------------------------------------------------------------------
// Host-side launch. Inputs (setup_inputs order):
//   0 query [4,128,128]  1 keys [4,512,128]  2 W1 [128,128]  3 b1 [128]
//   4 W2 [128,128]       5 b2 [128]          6 V [128,1]     7 bV [1]
// Output: [4,128,512] f32. Workspace: enc (1 MB) + dec (256 KB).
// ---------------------------------------------------------------------------
extern "C" void kernel_launch(void* const* d_in, const int* in_sizes, int n_in,
                              void* d_out, int out_size, void* d_ws, size_t ws_size,
                              hipStream_t stream)
{
    (void)in_sizes; (void)n_in; (void)out_size; (void)ws_size;

    const float* query = (const float*)d_in[0];
    const float* keys  = (const float*)d_in[1];
    const float* W1    = (const float*)d_in[2];
    const float* b1    = (const float*)d_in[3];
    const float* W2    = (const float*)d_in[4];
    const float* b2    = (const float*)d_in[5];
    const float* V     = (const float*)d_in[6];
    const float* bV    = (const float*)d_in[7];
    float* out = (float*)d_out;

    float* enc = (float*)d_ws;                            // [B*S, U] = [2048,128]
    float* dec = enc + (size_t)BATCH * SK * UNITS;        // [B*T, U] = [512,128]

    // enc: M=2048 -> 128*8=1024 tiles -> 128 blocks of 8 waves (exact).
    gemm_bias_wmma<<<dim3(128), dim3(256), 0, stream>>>(keys,  W1, b1, enc, BATCH * SK);
    // dec: M=512 -> 256 tiles -> 32 blocks (exact).
    gemm_bias_wmma<<<dim3(32),  dim3(256), 0, stream>>>(query, W2, b2, dec, BATCH * TQ);

    // scores: grid (S/16, T/16, B) = (32, 8, 4) = 1024 blocks.
    score_tanh_kernel<<<dim3(SK / 16, TQ / 16, BATCH), dim3(256), 0, stream>>>(
        enc, dec, V, bV, out);
}
